// temporal_2_1443109011522
// MI455X (gfx1250) — matrix operand order repaired and checked by reading, not compiled, not run on hardware
//
#include <hip/hip_runtime.h>
#include <hip/hip_bf16.h>

// ---------------------------------------------------------------------------
// CDNA5 (gfx1250) wave32 WMMA implementation of lag-attention.
// All matrix math runs through v_wmma_f32_16x16x32_bf16; all f32->bf16
// conversion is hoisted into a one-shot vectorized convert kernel.
// ---------------------------------------------------------------------------

typedef __attribute__((ext_vector_type(16))) __bf16         v16bf;
typedef __attribute__((ext_vector_type(2)))  __bf16         v2bf;
typedef __attribute__((ext_vector_type(8)))  float          v8f;
typedef __attribute__((ext_vector_type(8)))  unsigned short us8;

union FragU { v16bf v; us8 h[2]; };

__device__ __forceinline__ unsigned short f2bf(float f) {
    union { float f; unsigned u; } x; x.f = f;
    unsigned r = x.u + 0x7FFFu + ((x.u >> 16) & 1u);   // round-to-nearest-even
    return (unsigned short)(r >> 16);
}

// Pack two f32 -> bf16x2; use hardware v_cvt_pk_bf16_f32 when available.
__device__ __forceinline__ unsigned pack2bf(float a, float b) {
#if __has_builtin(__builtin_amdgcn_cvt_pk_bf16_f32)
    union { v2bf v; unsigned u; } x;
    x.v = __builtin_amdgcn_cvt_pk_bf16_f32(a, b);
    return x.u;
#else
    return (unsigned)f2bf(a) | ((unsigned)f2bf(b) << 16);
#endif
}

__device__ __forceinline__ void storeC(float* p, float v)          { *p = v; }
__device__ __forceinline__ void storeC(unsigned short* p, float v) { *p = f2bf(v); }

// A-matrix fragment (16x32 bf16, MxK) from LDS laid out row-major [row][k].
// ISA layout: lanes 0-15 -> M=lane, K pairs {0..7,16..23}; lanes 16-31 -> {8..15,24..31}.
__device__ __forceinline__ v16bf load_a_frag(const unsigned short* lds, int stride,
                                             int row0, int k0) {
    int lane = threadIdx.x & 31;
    int lo = lane & 15, hi = lane >> 4;
    const unsigned short* p = lds + (row0 + lo) * stride + k0 + hi * 8;
    FragU f;
    f.h[0] = *(const us8*)(p);        // K = kb .. kb+7
    f.h[1] = *(const us8*)(p + 16);   // K = kb+16 .. kb+23
    return f.v;
}

// B-matrix fragment (32x16 bf16, KxN) from LDS staged TRANSPOSED: [n][k] contiguous in k.
// ISA layout: lanes 0-15 -> N=lane, K {0..7,16..23}; lanes 16-31 -> {8..15,24..31}.
__device__ __forceinline__ v16bf load_b_frag(const unsigned short* lds, int stride,
                                             int col0, int k0) {
    int lane = threadIdx.x & 31;
    int lo = lane & 15, hi = lane >> 4;
    const unsigned short* p = lds + (col0 + lo) * stride + k0 + hi * 8;
    FragU f;
    f.h[0] = *(const us8*)(p);
    f.h[1] = *(const us8*)(p + 16);
    return f.v;
}

__device__ __forceinline__ v8f wmma_bf16(v16bf a, v16bf b, v8f c) {
    return __builtin_amdgcn_wmma_f32_16x16x32_bf16(false, a, false, b,
                                                   (short)0, c, false, false);
}

// ---------------------------------------------------------------------------
// One-shot f32 -> bf16 conversion, 4 floats / thread (b128 load, b64 store).
// ---------------------------------------------------------------------------
__global__ __launch_bounds__(256)
void cvt_f32_bf16(const float* __restrict__ src, unsigned* __restrict__ dst, long n4) {
    long i = (long)blockIdx.x * blockDim.x + threadIdx.x;
    if (i >= n4) return;
    float4 f = ((const float4*)src)[i];
    uint2 o;
    o.x = pack2bf(f.x, f.y);
    o.y = pack2bf(f.z, f.w);
    ((uint2*)dst)[i] = o;
}

// ---------------------------------------------------------------------------
// Generic bf16 GEMM:  C[M,N] = remap(A)[M,K] * B[K,N] + bias[N]
// A, B are bf16 (pre-converted); bias f32; C is f32 or bf16.
// REMAP==1 selects input[:, -1]: global row g -> A row (g/128)*640 + 512 + (g%128)
// Block: 256 threads / 8 waves, 128x128 tile; wave (wr,wc) owns 32x64.
// ---------------------------------------------------------------------------
template <int REMAP, typename TC>
__global__ __launch_bounds__(256)
void gemm_bias(const unsigned short* __restrict__ A,
               const unsigned short* __restrict__ B,
               const float* __restrict__ bias, TC* __restrict__ C,
               int N, int K) {
    __shared__ __align__(16) unsigned short As[128 * 32];
    __shared__ __align__(16) unsigned short Bs[128 * 32];

    const int n0 = blockIdx.x * 128;
    const int m0 = blockIdx.y * 128;
    const int t  = threadIdx.x;
    const int w  = t >> 5;
    const int wr = w >> 1;       // 0..3 : 32-row group
    const int wc = w & 1;        // 0..1 : 64-col group

    v8f acc[2][4];
    #pragma unroll
    for (int i = 0; i < 2; ++i)
        #pragma unroll
        for (int j = 0; j < 4; ++j) acc[i][j] = (v8f){};

    const int srow = t >> 1;               // 0..127
    const int scol = (t & 1) * 16;         // 0 / 16
    const int bkk  = t & 31;               // 0..31
    const int bn   = (t >> 5) * 16;        // 0..112

    long arow;
    {
        int g = m0 + srow;
        arow = REMAP ? (long)((g >> 7) * 640 + 512 + (g & 127)) : (long)g;
    }

    for (int kk = 0; kk < K; kk += 32) {
        __syncthreads();
        // stage A tile (128 x 32): straight bf16 copy, two b128 each
        {
            const us8* src = (const us8*)(A + arow * K + kk + scol);
            us8* dst = (us8*)(As + srow * 32 + scol);
            if (kk + 32 < K) __builtin_prefetch((const unsigned short*)src + 32, 0, 1);
            dst[0] = src[0];
            dst[1] = src[1];
        }
        // stage B tile (32 x 128) TRANSPOSED [n][k]: vector read, scatter write
        {
            const unsigned short* src = B + (long)(kk + bkk) * N + n0 + bn;
            us8 r0 = *(const us8*)(src);
            us8 r1 = *(const us8*)(src + 8);
            #pragma unroll
            for (int i = 0; i < 8; ++i) {
                Bs[(bn + i)     * 32 + bkk] = r0[i];
                Bs[(bn + 8 + i) * 32 + bkk] = r1[i];
            }
        }
        __syncthreads();

        v16bf bfrag[4];
        #pragma unroll
        for (int ni = 0; ni < 4; ++ni)
            bfrag[ni] = load_b_frag(Bs, 32, wc * 64 + ni * 16, 0);
        #pragma unroll
        for (int mi = 0; mi < 2; ++mi) {
            v16bf afrag = load_a_frag(As, 32, wr * 32 + mi * 16, 0);
            #pragma unroll
            for (int ni = 0; ni < 4; ++ni)
                acc[mi][ni] = wmma_bf16(afrag, bfrag[ni], acc[mi][ni]);
        }
    }

    // C/D layout: VGPR r -> M = r + 8*(lane>>4); N = lane&15
    const int lane = t & 31, lo = lane & 15, hi = lane >> 4;
    #pragma unroll
    for (int mi = 0; mi < 2; ++mi)
        #pragma unroll
        for (int ni = 0; ni < 4; ++ni) {
            int col = n0 + wc * 64 + ni * 16 + lo;
            float bv = bias[col];
            #pragma unroll
            for (int r = 0; r < 8; ++r) {
                int row = m0 + wr * 32 + mi * 16 + r + hi * 8;
                storeC(C + (long)row * N + col, acc[mi][ni][r] + bv);
            }
        }
}

// ---------------------------------------------------------------------------
// Fused attention per (b, lag, head): S = QK^T/8, causal, softmax, P*V.
// Q,K,V are bf16 projections: Q/V (b*128, 512), K (b*5*128, 512).
// inter written bf16 in (b, s, l, h, dh) order for the output GEMM.
// ---------------------------------------------------------------------------
__global__ __launch_bounds__(256)
void attention_kernel(const unsigned short* __restrict__ Q,
                      const unsigned short* __restrict__ Kp,
                      const unsigned short* __restrict__ V,
                      unsigned short* __restrict__ inter) {
    const int blk = blockIdx.x;      // b*40 + l*8 + h
    const int h = blk & 7;
    const int l = (blk >> 3) % 5;
    const int b = blk / 40;

    // 48KB LDS: [Qs 128x64 | Ks 128x64] overlaid later by Ps 128x128; Vs 64x128.
    __shared__ __align__(16) unsigned short smem[24576];
    unsigned short* Qs = smem;                 // 8192
    unsigned short* Ks = smem + 8192;          // 8192
    unsigned short* Vs = smem + 16384;         // 8192 (transposed [d][j])
    unsigned short* Ps = smem;                 // 16384 (overlays Qs+Ks)

    const int t = threadIdx.x;
    // ---- stage Q, K row-major and V transposed ----
    {
        int i  = t >> 1;
        int cb = (t & 1) * 32;
        const unsigned short* qsrc = Q  + (long)(b * 128 + i) * 512 + h * 64 + cb;
        const unsigned short* ksrc = Kp + (long)(((b * 5 + l) * 128) + i) * 512 + h * 64 + cb;
        const unsigned short* vsrc = V  + (long)(b * 128 + i) * 512 + h * 64 + cb;
        #pragma unroll
        for (int j2 = 0; j2 < 4; ++j2) {     // vectorized Q/K copies
            ((us8*)(Qs + i * 64 + cb))[j2] = ((const us8*)qsrc)[j2];
            ((us8*)(Ks + i * 64 + cb))[j2] = ((const us8*)ksrc)[j2];
        }
        #pragma unroll
        for (int j = 0; j < 32; ++j)
            Vs[(cb + j) * 128 + i] = vsrc[j];     // transpose scatter
    }
    __syncthreads();

    const int w = t >> 5, lane = t & 31, lo = lane & 15, hi = lane >> 4;

    // ---- scores: wave w owns rows w*16..w*16+15, all 128 cols ----
    v8f sc[8];
    #pragma unroll
    for (int ct = 0; ct < 8; ++ct) sc[ct] = (v8f){};
    #pragma unroll
    for (int kk = 0; kk < 64; kk += 32) {
        v16bf af = load_a_frag(Qs, 64, w * 16, kk);
        #pragma unroll
        for (int ct = 0; ct < 8; ++ct) {
            v16bf bf = load_b_frag(Ks, 64, ct * 16, kk);
            sc[ct] = wmma_bf16(af, bf, sc[ct]);
        }
    }

    // ---- scale + causal mask + softmax (row lives in one 16-lane half) ----
    #pragma unroll
    for (int r = 0; r < 8; ++r) {
        const int i = w * 16 + r + hi * 8;
        float vv[8];
        float m = -INFINITY;
        #pragma unroll
        for (int ct = 0; ct < 8; ++ct) {
            int j = ct * 16 + lo;
            float s = sc[ct][r] * 0.125f;          // 1/sqrt(64)
            if (j > i) s = -INFINITY;
            vv[ct] = s;
            m = fmaxf(m, s);
        }
        #pragma unroll
        for (int off = 1; off < 16; off <<= 1) m = fmaxf(m, __shfl_xor(m, off, 32));
        float sum = 0.f;
        #pragma unroll
        for (int ct = 0; ct < 8; ++ct) { float e = __expf(vv[ct] - m); vv[ct] = e; sum += e; }
        #pragma unroll
        for (int off = 1; off < 16; off <<= 1) sum += __shfl_xor(sum, off, 32);
        float inv = 1.0f / sum;
        #pragma unroll
        for (int ct = 0; ct < 8; ++ct) sc[ct][r] = vv[ct] * inv;
    }

    __syncthreads();   // everyone done reading Qs/Ks before Ps overlays them
    #pragma unroll
    for (int ct = 0; ct < 8; ++ct)
        #pragma unroll
        for (int r = 0; r < 8; ++r) {
            int i = w * 16 + r + hi * 8;
            Ps[i * 128 + ct * 16 + lo] = f2bf(sc[ct][r]);
        }
    __syncthreads();

    // ---- P (128x128) * V (128x64): wave w owns rows w*16, 4 col tiles ----
    v8f o[4];
    #pragma unroll
    for (int ni = 0; ni < 4; ++ni) o[ni] = (v8f){};
    #pragma unroll
    for (int kk = 0; kk < 128; kk += 32) {
        v16bf af = load_a_frag(Ps, 128, w * 16, kk);
        #pragma unroll
        for (int ni = 0; ni < 4; ++ni) {
            v16bf bf = load_b_frag(Vs, 128, ni * 16, kk);
            o[ni] = wmma_bf16(af, bf, o[ni]);
        }
    }

    // inter[(b,s,l,h,dh)] bf16, row length 2560
    #pragma unroll
    for (int ni = 0; ni < 4; ++ni) {
        int d = ni * 16 + lo;
        #pragma unroll
        for (int r = 0; r < 8; ++r) {
            int i = w * 16 + r + hi * 8;
            inter[((long)(b * 128 + i) * 5 + l) * 512 + h * 64 + d] = f2bf(o[ni][r]);
        }
    }
}

// ---------------------------------------------------------------------------
extern "C" void kernel_launch(void* const* d_in, const int* in_sizes, int n_in,
                              void* d_out, int out_size, void* d_ws, size_t ws_size,
                              hipStream_t stream) {
    const float* input = (const float*)d_in[0];
    const float* Wq = (const float*)d_in[1];
    const float* bq = (const float*)d_in[2];
    const float* Wk = (const float*)d_in[3];
    const float* bk = (const float*)d_in[4];
    const float* Wv = (const float*)d_in[5];
    const float* bv = (const float*)d_in[6];
    const float* Wo = (const float*)d_in[7];
    const float* bo = (const float*)d_in[8];
    float* out = (float*)d_out;

    // workspace partition (bf16 elements)
    const size_t IN_E  = 40960UL * 512;    // full input, all lags
    const size_t W_E   = 512UL * 512;      // Wq/Wk/Wv
    const size_t WO_E  = 2560UL * 512;     // Wo
    const size_t QV_E  = 8192UL * 512;     // Q / V projections
    const size_t K_E   = 40960UL * 512;    // K projection
    const size_t I_E   = 8192UL * 2560;    // inter
    const size_t need = (IN_E + 3 * W_E + WO_E + 2 * QV_E + K_E + I_E) * 2;
    if (ws_size < need) return;

    unsigned short* bfIn = (unsigned short*)d_ws;
    unsigned short* bfWq = bfIn + IN_E;
    unsigned short* bfWk = bfWq + W_E;
    unsigned short* bfWv = bfWk + W_E;
    unsigned short* bfWo = bfWv + W_E;
    unsigned short* wsQ  = bfWo + WO_E;
    unsigned short* wsV  = wsQ + QV_E;
    unsigned short* wsK  = wsV + QV_E;
    unsigned short* wsI  = wsK + K_E;

    dim3 blk(256);
    auto cvt = [&](const float* s, unsigned short* d, size_t n) {
        long n4 = (long)(n / 4);
        cvt_f32_bf16<<<dim3((unsigned)((n4 + 255) / 256)), blk, 0, stream>>>(
            s, (unsigned*)d, n4);
    };

    // one-shot conversions to bf16
    cvt(input, bfIn, IN_E);
    cvt(Wq, bfWq, W_E);
    cvt(Wk, bfWk, W_E);
    cvt(Wv, bfWv, W_E);
    cvt(Wo, bfWo, WO_E);

    // Q = last @ Wq + bq   (REMAP=1 selects input[:, -1])
    gemm_bias<1, unsigned short><<<dim3(4, 64), blk, 0, stream>>>(
        bfIn, bfWq, bq, wsQ, 512, 512);
    // V = last @ Wv + bv
    gemm_bias<1, unsigned short><<<dim3(4, 64), blk, 0, stream>>>(
        bfIn, bfWv, bv, wsV, 512, 512);
    // K = input @ Wk + bk  over all lags
    gemm_bias<0, unsigned short><<<dim3(4, 320), blk, 0, stream>>>(
        bfIn, bfWk, bk, wsK, 512, 512);

    // fused attention per (b, lag, head)
    attention_kernel<<<dim3(64 * 5 * 8), blk, 0, stream>>>(wsQ, wsK, wsV, wsI);

    // out = inter @ Wo + bo
    gemm_bias<0, float><<<dim3(4, 64), blk, 0, stream>>>(
        wsI, bfWo, bo, out, 512, 2560);

    (void)in_sizes; (void)n_in; (void)out_size;
}
